// MatrixLSTMCell_4733053960556
// MI455X (gfx1250) — matrix operand order repaired and checked by reading, not compiled, not run on hardware
//
#include <hip/hip_runtime.h>
#include <cmath>

#define DIMX 384
#define NHX 6
#define DHX 64
#define BATCH 2
#define SEQ 2048
#define BM 128
#define BN 64
#define DHP (DHX + 8)   // padded LDS row (keeps 16B alignment, breaks conflicts)
#define BNP (BN + 8)

typedef __attribute__((ext_vector_type(16))) __bf16 v16bf;
typedef __attribute__((ext_vector_type(8)))  __bf16 v8bf;
typedef __attribute__((ext_vector_type(2)))  __bf16 v2bf;
typedef __attribute__((ext_vector_type(8)))  float  v8f;

static __device__ __forceinline__ v8f wmma_bf16(v16bf a, v16bf b, v8f c) {
  // D = A(16x32) * B(32x16) + C, fp32 accumulate
  return __builtin_amdgcn_wmma_f32_16x16x32_bf16(false, a, false, b, (short)0, c, false, false);
}

static __device__ __forceinline__ v8f zero8() {
  v8f z = {0.f, 0.f, 0.f, 0.f, 0.f, 0.f, 0.f, 0.f};
  return z;
}

// ---------------------------------------------------------------------------
// Kernel 1: gate pre-activations  ig/fg[b,h,s] = concat(q,k,v)[b,s] . w[h] + bias
// ---------------------------------------------------------------------------
__global__ __launch_bounds__(32)
void mlstm_gates_kernel(const float* __restrict__ q, const float* __restrict__ k,
                        const float* __restrict__ v,
                        const float* __restrict__ igw, const float* __restrict__ igb,
                        const float* __restrict__ fgw, const float* __restrict__ fgb,
                        float* __restrict__ igArr, float* __restrict__ fgArr) {
  const int bs = blockIdx.x;
  const int b = bs / SEQ;
  const int s = bs % SEQ;
  const int lane = threadIdx.x;
  const float* qr = q + (size_t)bs * DIMX;
  const float* kr = k + (size_t)bs * DIMX;
  const float* vr = v + (size_t)bs * DIMX;

  float accI[NHX], accF[NHX];
#pragma unroll
  for (int h = 0; h < NHX; ++h) { accI[h] = 0.f; accF[h] = 0.f; }

  for (int c = lane; c < 3 * DIMX; c += 32) {
    float x = (c < DIMX) ? qr[c] : ((c < 2 * DIMX) ? kr[c - DIMX] : vr[c - 2 * DIMX]);
#pragma unroll
    for (int h = 0; h < NHX; ++h) {
      accI[h] = fmaf(x, igw[h * 3 * DIMX + c], accI[h]);
      accF[h] = fmaf(x, fgw[h * 3 * DIMX + c], accF[h]);
    }
  }
#pragma unroll
  for (int off = 16; off > 0; off >>= 1) {
#pragma unroll
    for (int h = 0; h < NHX; ++h) {
      accI[h] += __shfl_down(accI[h], off, 32);
      accF[h] += __shfl_down(accF[h], off, 32);
    }
  }
  if (lane == 0) {
#pragma unroll
    for (int h = 0; h < NHX; ++h) {
      igArr[(size_t)(b * NHX + h) * SEQ + s] = accI[h] + igb[h];
      fgArr[(size_t)(b * NHX + h) * SEQ + s] = accF[h] + fgb[h];
    }
  }
}

// ---------------------------------------------------------------------------
// Kernel 2: per-(b,h) scan.  cum = cumsum(log_sigmoid(fg)); a = ig - cum;
// A = prefix-max(a); elow = exp(-(cum + A))  (= exp(-max_log_D))
// ---------------------------------------------------------------------------
__global__ __launch_bounds__(32)
void mlstm_scan_kernel(const float* __restrict__ igArr, const float* __restrict__ fgArr,
                       float* __restrict__ aArr, float* __restrict__ AmArr,
                       float* __restrict__ elArr) {
  const size_t base = (size_t)blockIdx.x * SEQ;
  const int lane = threadIdx.x;
  float csum = 0.f;
  float cmax = -INFINITY;
  for (int c0 = 0; c0 < SEQ; c0 += 32) {
    const int s = c0 + lane;
    const float fgv = fgArr[base + s];
    const float lf = fminf(fgv, 0.f) - log1pf(__expf(-fabsf(fgv)));
    float ps = lf;
#pragma unroll
    for (int d = 1; d < 32; d <<= 1) {
      float o = __shfl_up(ps, d, 32);
      if (lane >= d) ps += o;
    }
    const float cum = csum + ps;
    const float a = igArr[base + s] - cum;
    float pm = a;
#pragma unroll
    for (int d = 1; d < 32; d <<= 1) {
      float o = __shfl_up(pm, d, 32);
      if (lane >= d) pm = fmaxf(pm, o);
    }
    const float Am = fmaxf(cmax, pm);
    aArr[base + s] = a;
    AmArr[base + s] = Am;
    elArr[base + s] = __expf(-(cum + Am));
    csum = __shfl(cum, 31, 32);
    cmax = __shfl(Am, 31, 32);
  }
}

// ---------------------------------------------------------------------------
// Kernel 3: main causal WMMA sweep.
// grid = B*NH*(SEQ/BM) blocks of 256 threads (8 waves); wave w owns 16 query rows.
// S^T = K Q^T (D-layout: lane = query col), gate C = (qk/8)*exp(a_j - A_i),
// causal mask, per-lane row sums, shfl_xor(16) repack of P^T into B-operands,
// h^T += V^T P^T.  Epilogue: normalizer + MultiHeadLayerNorm, b128 stores.
// ---------------------------------------------------------------------------
__global__ __launch_bounds__(256)
void mlstm_attn_kernel(const float* __restrict__ q, const float* __restrict__ k,
                       const float* __restrict__ v,
                       const float* __restrict__ aArr, const float* __restrict__ AmArr,
                       const float* __restrict__ elArr,
                       const float* __restrict__ wOut, float* __restrict__ out) {
  __shared__ __bf16 sQ[BM][DHP];   // row-major queries (bf16)
  __shared__ __bf16 sK[BN][DHP];   // row-major keys
  __shared__ __bf16 sVT[DHX][BNP]; // transposed values
  __shared__ float sA[BN];

  const int blk = blockIdx.x;
  const int mblk = blk % (SEQ / BM);
  const int bh = blk / (SEQ / BM);
  const int h = bh % NHX;
  const int b = bh / NHX;
  const int row0 = mblk * BM;
  const int tid = threadIdx.x;
  const int lane = tid & 31;
  const int wave = tid >> 5;
  const bool hiHalf = (lane >= 16);
  const int qcol = lane & 15;

  const float* qg = q + (size_t)b * SEQ * DIMX + h * DHX;
  const float* kg = k + (size_t)b * SEQ * DIMX + h * DHX;
  const float* vg = v + (size_t)b * SEQ * DIMX + h * DHX;
  const size_t gbase = (size_t)(b * NHX + h) * SEQ;

  // stage Q tile (fp32 -> bf16), float2 loads + bf16x2 stores
#pragma unroll
  for (int it = 0; it < (BM * DHX) / (2 * 256); ++it) {
    const int e = it * 256 + tid;           // [0, 4096)
    const int r = e >> 5, c0 = (e & 31) * 2;
    const float2 xv = *(const float2*)&qg[(size_t)(row0 + r) * DIMX + c0];
    v2bf p = {(__bf16)xv.x, (__bf16)xv.y};
    *(v2bf*)&sQ[r][c0] = p;
  }
  __syncthreads();

  const int qrow = row0 + wave * 16 + qcol;       // this lane's query row
  const float A_i8 = AmArr[gbase + qrow] + 2.0794415416798357f;  // + ln(8): folds 1/sqrt(DH)
  const float elow_i = elArr[gbase + qrow];       // exp(-max_log_D_i)
  const int imax_s = __builtin_amdgcn_readfirstlane(row0 + wave * 16 + 15);

  // Q^T as WMMA B operand: lane -> query column, K = features
  const int qfo = hiHalf ? 16 : 0;
  const v16bf qb0 = *(const v16bf*)&sQ[wave * 16 + qcol][qfo];
  const v16bf qb1 = *(const v16bf*)&sQ[wave * 16 + qcol][32 + qfo];

  v8f acc[4];
#pragma unroll
  for (int t = 0; t < 4; ++t) acc[t] = zero8();
  float rowsum = 0.f;

  const int fo = hiHalf ? 8 : 0;
  const int nk = row0 + BM;

  for (int j0 = 0; j0 < nk; j0 += BN) {
    __syncthreads();
    // stage K rows (float2 -> bf16x2)
#pragma unroll
    for (int it = 0; it < (BN * DHX) / (2 * 256); ++it) {
      const int e = it * 256 + tid;         // [0, 2048)
      const int r = e >> 5, c0 = (e & 31) * 2;
      const float2 xv = *(const float2*)&kg[(size_t)(j0 + r) * DIMX + c0];
      v2bf p = {(__bf16)xv.x, (__bf16)xv.y};
      *(v2bf*)&sK[r][c0] = p;
    }
    // stage V transposed: pack key pairs -> b32 stores
#pragma unroll
    for (int it = 0; it < (BN * DHX) / (2 * 256); ++it) {
      const int e = it * 256 + tid;         // [0, 2048)
      const int c = e & 63;                 // feature
      const int r0 = (e >> 6) * 2;          // key pair
      const float x0 = vg[(size_t)(j0 + r0) * DIMX + c];
      const float x1 = vg[(size_t)(j0 + r0 + 1) * DIMX + c];
      v2bf p = {(__bf16)x0, (__bf16)x1};
      *(v2bf*)&sVT[c][r0] = p;
    }
    if (tid < BN) sA[tid] = aArr[gbase + j0 + tid];
    __syncthreads();

    // prefetch next K/V tile into near caches while this tile computes
    // (locality 3 -> WGP scope: pull into all cache levels on miss)
    if (j0 + BN < nk && tid < 128) {
      const int pr = tid >> 1, pc = (tid & 1) * 32;
      __builtin_prefetch(&kg[(size_t)(j0 + BN + pr) * DIMX + pc], 0, 3);
      __builtin_prefetch(&vg[(size_t)(j0 + BN + pr) * DIMX + pc], 0, 3);
    }

    float st[4][8];  // four 16-key S^T subtiles (D layout), gated/masked
#pragma unroll
    for (int kt = 0; kt < 4; ++kt) {
      const int jb = j0 + kt * 16;
      if (jb <= imax_s) {   // scalar branch (SGPR compare)
        const int krow = kt * 16 + qcol;
        v16bf ka0, ka1;
        ((v8bf*)&ka0)[0] = *(const v8bf*)&sK[krow][qfo];
        ((v8bf*)&ka0)[1] = *(const v8bf*)&sK[krow][8 + qfo];
        ((v8bf*)&ka1)[0] = *(const v8bf*)&sK[krow][32 + qfo];
        ((v8bf*)&ka1)[1] = *(const v8bf*)&sK[krow][40 + qfo];
        v8f sacc = zero8();
        sacc = wmma_bf16(ka0, qb0, sacc);
        sacc = wmma_bf16(ka1, qb1, sacc);
        // contiguous gate values for this lane's 8 elements
        float aj[8];
        *(float4*)&aj[0] = *(const float4*)&sA[kt * 16 + fo];
        *(float4*)&aj[4] = *(const float4*)&sA[kt * 16 + fo + 4];
#pragma unroll
        for (int r = 0; r < 8; ++r) {
          const int j = jb + r + fo;                       // key index
          const float w = __expf(aj[r] - A_i8);            // exp(a_j - A_i)/8
          const float pv = (j <= qrow) ? sacc[r] * w : 0.f;  // branchless cndmask
          st[kt][r] = pv;
          rowsum += pv;
        }
      } else {
#pragma unroll
        for (int r = 0; r < 8; ++r) st[kt][r] = 0.f;
      }
    }

    // repack P^T (D layout) -> WMMA B operands and accumulate h^T += V^T P^T
#pragma unroll
    for (int pair = 0; pair < 2; ++pair) {
      if (j0 + pair * 32 <= imax_s) {   // scalar branch
        float sx0[8], sx1[8];
#pragma unroll
        for (int r = 0; r < 8; ++r) {
          sx0[r] = __shfl_xor(st[2 * pair][r], 16, 32);
          sx1[r] = __shfl_xor(st[2 * pair + 1][r], 16, 32);
        }
        v16bf pb;
#pragma unroll
        for (int vv = 0; vv < 4; ++vv) {
          pb[2 * vv] = (__bf16)(hiHalf ? sx1[2 * vv] : st[2 * pair][2 * vv]);
          pb[2 * vv + 1] = (__bf16)(hiHalf ? sx1[2 * vv + 1] : st[2 * pair][2 * vv + 1]);
        }
#pragma unroll
        for (int vv = 4; vv < 8; ++vv) {
          pb[2 * vv] = (__bf16)(hiHalf ? st[2 * pair + 1][2 * vv - 8] : sx0[2 * vv - 8]);
          pb[2 * vv + 1] = (__bf16)(hiHalf ? st[2 * pair + 1][2 * vv - 7] : sx0[2 * vv - 7]);
        }
#pragma unroll
        for (int ft = 0; ft < 4; ++ft) {
          const int frow = ft * 16 + qcol;
          const int ko = pair * 32 + qfo;
          v16bf va;
          ((v8bf*)&va)[0] = *(const v8bf*)&sVT[frow][ko];
          ((v8bf*)&va)[1] = *(const v8bf*)&sVT[frow][ko + 8];
          acc[ft] = wmma_bf16(va, pb, acc[ft]);
        }
      }
    }
  }

  // ---- epilogue: normalizer + per-head LayerNorm + store ----
  const float rs = rowsum + __shfl_xor(rowsum, 16, 32);
  const float norm = fmaxf(fabsf(rs), elow_i) + 1e-6f;
  const float inv = 1.0f / norm;

  float hv[4][8];
  float sum = 0.f, ssq = 0.f;
#pragma unroll
  for (int t = 0; t < 4; ++t)
#pragma unroll
    for (int r = 0; r < 8; ++r) {
      const float x = acc[t][r] * inv;
      hv[t][r] = x;
      sum += x;
      ssq += x * x;
    }
  sum += __shfl_xor(sum, 16, 32);
  ssq += __shfl_xor(ssq, 16, 32);
  const float mean = sum * (1.f / 64.f);
  const float var = ssq * (1.f / 64.f) - mean * mean;
  const float rstd = rsqrtf(var + 1e-5f);

  float* og = out + ((size_t)(b * SEQ) + qrow) * DIMX + h * DHX;
  const float* wb = wOut + h * DHX;
  const int fo2 = hiHalf ? 8 : 0;
#pragma unroll
  for (int t = 0; t < 4; ++t) {
    const int d0 = t * 16 + fo2;
    float4 lo4, hi4;
    lo4.x = (hv[t][0] - mean) * rstd * (1.f + wb[d0 + 0]);
    lo4.y = (hv[t][1] - mean) * rstd * (1.f + wb[d0 + 1]);
    lo4.z = (hv[t][2] - mean) * rstd * (1.f + wb[d0 + 2]);
    lo4.w = (hv[t][3] - mean) * rstd * (1.f + wb[d0 + 3]);
    hi4.x = (hv[t][4] - mean) * rstd * (1.f + wb[d0 + 4]);
    hi4.y = (hv[t][5] - mean) * rstd * (1.f + wb[d0 + 5]);
    hi4.z = (hv[t][6] - mean) * rstd * (1.f + wb[d0 + 6]);
    hi4.w = (hv[t][7] - mean) * rstd * (1.f + wb[d0 + 7]);
    *(float4*)(og + d0) = lo4;
    *(float4*)(og + d0 + 4) = hi4;
  }
}

// ---------------------------------------------------------------------------
extern "C" void kernel_launch(void* const* d_in, const int* in_sizes, int n_in,
                              void* d_out, int out_size, void* d_ws, size_t ws_size,
                              hipStream_t stream) {
  const float* q = (const float*)d_in[0];
  const float* k = (const float*)d_in[1];
  const float* v = (const float*)d_in[2];
  const float* igw = (const float*)d_in[3];
  const float* igb = (const float*)d_in[4];
  const float* fgw = (const float*)d_in[5];
  const float* fgb = (const float*)d_in[6];
  const float* wOut = (const float*)d_in[7];
  float* out = (float*)d_out;

  float* ws = (float*)d_ws;
  const size_t N = (size_t)BATCH * NHX * SEQ;  // 24576
  float* igArr = ws;
  float* fgArr = igArr + N;
  float* aArr = fgArr + N;
  float* AmArr = aArr + N;
  float* elArr = AmArr + N;

  mlstm_gates_kernel<<<BATCH * SEQ, 32, 0, stream>>>(q, k, v, igw, igb, fgw, fgb,
                                                     igArr, fgArr);
  mlstm_scan_kernel<<<BATCH * NHX, 32, 0, stream>>>(igArr, fgArr, aArr, AmArr, elArr);
  mlstm_attn_kernel<<<BATCH * NHX * (SEQ / BM), 256, 0, stream>>>(
      q, k, v, aArr, AmArr, elArr, wOut, out);
}
